// Node2Prop2_13829794693352
// MI455X (gfx1250) — matrix operand order repaired and checked by reading, not compiled, not run on hardware
//
#include <hip/hip_runtime.h>

typedef __attribute__((ext_vector_type(16))) __bf16 v16bf;
typedef __attribute__((ext_vector_type(8)))  float  v8f;
typedef __attribute__((ext_vector_type(16))) float  v16f;

#define IN_DIM 128
#define HID 128
#define STRIPS 4
#define ROWS_PER_BLOCK (16 * 8 * STRIPS)   // 512 nodes per block
#define LOG2F_CONST 0.6931471805599453f

__global__ void zero_out_kernel(float* __restrict__ out, int n) {
    int i = blockIdx.x * 256 + threadIdx.x;
    if (i < n) out[i] = 0.0f;
}

__global__ __launch_bounds__(256) void node2prop2_kernel(
    const float* __restrict__ x,      // (N, 128)
    const float* __restrict__ W1,     // (128, 128) row-major (k, n)
    const float* __restrict__ b1,     // (128,)
    const float* __restrict__ W2,     // (128, 1)
    const int*   __restrict__ batch,  // (N,) sorted
    float* __restrict__ out,          // (n_graphs,)
    int nNodes)
{
    // W1 staged as bf16 WMMA-B fragments: 32 tiles (kt 0..3, nt 0..7),
    // each tile = 32 lanes x 16 bf16 contiguous per lane = 1 KB.
    __shared__ __align__(32) unsigned short ldsB[32 * 512]; // 32 KB
    __shared__ float b1s[HID];
    __shared__ float w2s[HID];

    const int tid  = threadIdx.x;
    const int lane = tid & 31;
    const int wave = tid >> 5;

    if (tid < HID)            b1s[tid]       = b1[tid];
    else if (tid < 2 * HID)   w2s[tid - HID] = W2[tid - HID];

    // Stage W1 into B-fragment layout (once per block, reused by all strips):
    // tile t = kt*8 + nt; lane l: N = nt*16 + (l&15), K = kt*32 + (l>>4)*16 + j
    for (int p = tid; p < 32 * 32; p += 256) {
        int t  = p >> 5;
        int l  = p & 31;
        int kt = t >> 3, nt = t & 7;
        int n  = nt * 16 + (l & 15);
        int kb = kt * 32 + (l >> 4) * 16;
        v16f tmp;
#pragma unroll
        for (int j = 0; j < 16; ++j)
            tmp[j] = W1[(kb + j) * HID + n];
        *(v16bf*)&ldsB[t * 512 + l * 16] = __builtin_convertvector(tmp, v16bf);
    }
    __syncthreads();

    const int ncol  = lane & 15;
    const int khalf = (lane >> 4) * 16;  // lanes 16-31 take the K+16 sub-block

    for (int s = 0; s < STRIPS; ++s) {
        // ---- A fragments: this wave owns 16 node-rows in this strip ----
        const int rowBase = blockIdx.x * ROWS_PER_BLOCK + (s * 8 + wave) * 16;
        int row  = rowBase + ncol;
        int rowC = row < nNodes ? row : (nNodes - 1);   // address clamp only
        const float* rp = x + (size_t)rowC * IN_DIM;

        v16bf a[4];
#pragma unroll
        for (int kt = 0; kt < 4; ++kt) {
            const float4* p0 = (const float4*)(rp + kt * 32 + khalf);
            float4 f0 = p0[0], f1 = p0[1];   // K = base+0..7
            float4 f2 = p0[2], f3 = p0[3];   // K = base+8..15
            v16f vals = { f0.x, f0.y, f0.z, f0.w, f1.x, f1.y, f1.z, f1.w,
                          f2.x, f2.y, f2.z, f2.w, f3.x, f3.y, f3.z, f3.w };
            a[kt] = __builtin_convertvector(vals, v16bf);
        }

        // ---- GEMM + fused bias/softplus/W2-dot ----
        float acc[8];
#pragma unroll
        for (int r = 0; r < 8; ++r) acc[r] = 0.0f;
        float wsum = 0.0f;

#pragma unroll
        for (int nt = 0; nt < 8; ++nt) {
            v8f c = {};
#pragma unroll
            for (int kt = 0; kt < 4; ++kt) {
                v16bf bfrag = *(const v16bf*)&ldsB[(kt * 8 + nt) * 512 + lane * 16];
                c = __builtin_amdgcn_wmma_f32_16x16x32_bf16(
                        false, a[kt], false, bfrag, (short)0, c, false, false);
            }
            const int N = nt * 16 + ncol;
            const float bb = b1s[N];
            const float ww = w2s[N];
            wsum += ww;
#pragma unroll
            for (int r = 0; r < 8; ++r) {
                float v = c[r] + bb;
                // stable softplus: max(v,0) + log1p(exp(-|v|)); ln2 shift folded out
                float p = fmaxf(v, 0.0f) + __logf(1.0f + __expf(-fabsf(v)));
                acc[r] += p * ww;
            }
        }

        // Reduce over N: butterfly within each 16-lane half
        // (C layout: lanes 0-15 hold M=r, lanes 16-31 hold M=r+8)
#pragma unroll
        for (int m = 1; m <= 8; m <<= 1) {
#pragma unroll
            for (int r = 0; r < 8; ++r)
                acc[r] += __shfl_xor(acc[r], m, 32);
            wsum += __shfl_xor(wsum, m, 32);
        }

        // lane 0 -> rows rowBase+0..7, lane 16 -> rows rowBase+8..15
        if ((lane & 15) == 0) {
            const int rbase = rowBase + (lane >> 4) * 8;
            const float shift = LOG2F_CONST * wsum;  // ln2 * sum(W2)
#pragma unroll
            for (int r = 0; r < 8; ++r) {
                int node = rbase + r;
                if (node < nNodes) {
                    int g = batch[node];
                    atomicAdd(&out[g], acc[r] - shift);
                }
            }
        }
    }
}

extern "C" void kernel_launch(void* const* d_in, const int* in_sizes, int n_in,
                              void* d_out, int out_size, void* d_ws, size_t ws_size,
                              hipStream_t stream) {
    const float* x     = (const float*)d_in[0];
    const float* W1    = (const float*)d_in[1];
    const float* b1    = (const float*)d_in[2];
    const float* W2    = (const float*)d_in[3];
    const int*   batch = (const int*)d_in[4];
    float* out = (float*)d_out;

    const int nNodes = in_sizes[0] / IN_DIM;

    // d_out is poisoned by the harness: zero it first (same stream => ordered)
    zero_out_kernel<<<(out_size + 255) / 256, 256, 0, stream>>>(out, out_size);

    const int nblk = (nNodes + ROWS_PER_BLOCK - 1) / ROWS_PER_BLOCK;
    node2prop2_kernel<<<nblk, 256, 0, stream>>>(x, W1, b1, W2, batch, out, nNodes);
}
